// AttentionWithRoPE_1408749273870
// MI455X (gfx1250) — matrix operand order repaired and checked by reading, not compiled, not run on hardware
//
#include <hip/hip_runtime.h>

typedef __attribute__((ext_vector_type(16))) __bf16   v16bf;
typedef __attribute__((ext_vector_type(8)))  float    v8f;
typedef __attribute__((ext_vector_type(4)))  unsigned u32x4;
typedef __attribute__((ext_vector_type(2)))  unsigned u32x2;
typedef __attribute__((ext_vector_type(4)))  float    f32x4;

constexpr int B  = 2, N = 2048, C = 1024, H = 16, Dh = 64;
constexpr int M  = B * N;          // 4096 rows
constexpr int JQ = 3 * C;          // 3072 qkv cols
constexpr float SCALE = 0.125f;    // Dh^-0.5

union Frag {                       // 32 bytes = 8 VGPRs
  v16bf    v;
  u32x4    q[2];
  unsigned u[8];
};

__device__ __forceinline__ unsigned short f32_to_bf16_bits(float f) {
  union { float f; unsigned u; } a; a.f = f;
  unsigned r = a.u + 0x7FFFu + ((a.u >> 16) & 1u);
  return (unsigned short)(r >> 16);
}
// pack two floats into one dword of 2x bf16 (RNE)
__device__ __forceinline__ unsigned pack_bf16x2(float lo, float hi) {
  union { float f; unsigned u; } a, b; a.f = lo; b.f = hi;
  unsigned ul = (a.u + 0x7FFFu + ((a.u >> 16) & 1u)) >> 16;
  unsigned uh = (b.u + 0x7FFFu + ((b.u >> 16) & 1u)) & 0xFFFF0000u;
  return ul | uh;
}
__device__ __forceinline__ v8f wmma_bf16(v16bf a, v16bf b, v8f c) {
  return __builtin_amdgcn_wmma_f32_16x16x32_bf16(false, a, false, b, (short)0, c,
                                                 false, false);
}

// ---------------------------------------------------------------------------
// Kernel 0: one-shot streaming f32 -> bf16 conversion (x, qkv_w, proj_w).
// Pure bandwidth; runs once so the GEMM inner loops carry zero convert VALU.
// ---------------------------------------------------------------------------
__global__ __launch_bounds__(256) void cvt_bf16_kernel(
    const float* __restrict__ src, unsigned short* __restrict__ dst, int n4)
{
  int i = blockIdx.x * 256 + threadIdx.x;
  if (i < n4) {
    f32x4 f = ((const f32x4*)src)[i];
    ((u32x2*)dst)[i] = (u32x2){pack_bf16x2(f.x, f.y), pack_bf16x2(f.z, f.w)};
  }
}

// ---------------------------------------------------------------------------
// Kernel 1: QKV GEMM (bf16 WMMA) fused with 2D RoPE on q,k.
// Wave owns a 32-aligned column pair so the RoPE partner (d, d+16) is in the
// sibling tile: the rotation is pure elementwise register math.
// ---------------------------------------------------------------------------
__global__ __launch_bounds__(128) void qkv_rope_kernel(
    const unsigned short* __restrict__ xb, const int* __restrict__ positions,
    const unsigned short* __restrict__ wb,
    unsigned short* __restrict__ q_ws, unsigned short* __restrict__ k_ws,
    unsigned short* __restrict__ v_ws)
{
  __shared__ unsigned short xt[16 * C];          // 32 KB bf16 X tile
  const int tid   = threadIdx.x;
  const int lane  = tid & 31;
  const int wave  = tid >> 5;
  const int mbase = blockIdx.x * 16;

  // cooperative stage: straight b128 copy of pre-converted X tile
  {
    const u32x4* src = (const u32x4*)(xb + (size_t)mbase * C);
    u32x4* dst = (u32x4*)xt;
    for (int i = tid; i < (16 * C) / 8; i += 128) dst[i] = src[i];
  }
  __syncthreads();

  const int jpair = blockIdx.y * 4 + wave;       // 0..95
  const int j0    = jpair * 32;
  const int n     = lane & 15;
  const int hh    = lane >> 4;

  const unsigned short* w0 = wb + (size_t)(j0 + n) * C;
  const unsigned short* w1 = wb + (size_t)(j0 + 16 + n) * C;

  v8f c0 = {}, c1 = {};
  for (int kc = 0; kc < C; kc += 32) {
    Frag a, b0, b1;
    const unsigned short* ar = &xt[n * C + kc + hh * 16];  // A: one 32B run
    a.q[0] = *(const u32x4*)ar;
    a.q[1] = *(const u32x4*)(ar + 8);
    const unsigned short* p0 = &w0[kc + hh * 16];          // B: one 32B run
    b0.q[0] = *(const u32x4*)p0;
    b0.q[1] = *(const u32x4*)(p0 + 8);
    const unsigned short* p1 = &w1[kc + hh * 16];
    b1.q[0] = *(const u32x4*)p1;
    b1.q[1] = *(const u32x4*)(p1 + 8);
    __builtin_prefetch(&w0[kc + 32], 0, 3);      // global_prefetch_b8
    c0 = wmma_bf16(a.v, b0.v, c0);
    c1 = wmma_bf16(a.v, b1.v, c1);
  }

  // epilogue: RoPE (q,k only) + scatter to [B,H,N,Dh] bf16
  const int which = j0 / C;                      // 0=q 1=k 2=v
  const int rem   = j0 % C;
  const int hIdx  = rem / Dh;
  const int dspan = rem % Dh;                    // 0 or 32
  const int comp  = dspan ? 1 : 0;
  const float inv = __expf(-(float)n * (4.605170185988092f / 16.0f)); // 100^(-n/16)

  unsigned short* dst = (which == 0) ? q_ws : (which == 1) ? k_ws : v_ws;

#pragma unroll
  for (int r = 0; r < 8; ++r) {
    int token = mbase + r + 8 * hh;              // C/D layout: row = r + 8*half
    int bb = token >> 11;
    int tt = token & (N - 1);
    float o0 = c0[r], o1 = c1[r];
    if (which < 2) {
      float pos = (float)positions[(size_t)token * 2 + comp];
      float ang = pos * inv;
      float s, cth;
      __sincosf(ang, &s, &cth);
      float x1 = o0, x2 = o1;
      o0 = x1 * cth - x2 * s;
      o1 = x1 * s + x2 * cth;
    }
    size_t base = (((size_t)(bb * H + hIdx) * N) + tt) * Dh;
    dst[base + dspan + n]      = f32_to_bf16_bits(o0);
    dst[base + dspan + 16 + n] = f32_to_bf16_bits(o1);
  }
}

// ---------------------------------------------------------------------------
// Kernel 2: flash attention. Block = 4 waves sharing one (b,h); K/V 64-key
// tiles cooperatively staged in LDS (V transposed), 16 v_wmma per block iter.
// ---------------------------------------------------------------------------
__global__ __launch_bounds__(128) void flash_attn_kernel(
    const unsigned short* __restrict__ q_ws,
    const unsigned short* __restrict__ k_ws,
    const unsigned short* __restrict__ v_ws,
    unsigned short* __restrict__ attn_ws)
{
  __shared__ unsigned short kt[64 * 64];          // 8 KB  [key][dh]
  __shared__ unsigned short vt[64 * 64];          // 8 KB  [dh][key] (transposed)
  __shared__ unsigned short p_lds[4][16 * 64];    // 8 KB  per-wave P scratch
  const int tid  = threadIdx.x, lane = tid & 31, wave = tid >> 5;
  const int qtile = blockIdx.x * 4 + wave;        // 0..127
  const int bh    = blockIdx.y;                   // 0..31
  const int h = bh & 15, b = bh >> 4;
  const int n = lane & 15, hh = lane >> 4;

  const size_t bh_base = (size_t)bh * N * Dh;
  const unsigned short* qp = q_ws + bh_base;
  const unsigned short* kp = k_ws + bh_base;
  const unsigned short* vp = v_ws + bh_base;
  unsigned short* pw = p_lds[wave];

  // Q fragments: A layout = two contiguous 16B runs per lane
  Frag qa[2];
#pragma unroll
  for (int cch = 0; cch < 2; ++cch) {
    const unsigned short* qr =
        &qp[(size_t)(qtile * 16 + n) * Dh + cch * 32 + hh * 16];
    qa[cch].q[0] = *(const u32x4*)qr;
    qa[cch].q[1] = *(const u32x4*)(qr + 8);
  }

  v8f o[4];
  v8f vzero = {};
#pragma unroll
  for (int dt = 0; dt < 4; ++dt) o[dt] = vzero;
  float mrun[8], lrun[8];
#pragma unroll
  for (int r = 0; r < 8; ++r) { mrun[r] = -1e30f; lrun[r] = 0.0f; }

  for (int kb = 0; kb < N; kb += 64) {
    __syncthreads();                              // protect kt/vt reuse
    // stage K tile: contiguous 8 KB, straight b128 copy
    {
      const u32x4* src = (const u32x4*)(kp + (size_t)kb * Dh);
      u32x4* dst = (u32x4*)kt;
      for (int i = tid; i < 512; i += 128) dst[i] = src[i];
    }
    // stage V tile transposed: [key][dh] -> [dh][key]
    {
      const unsigned* src = (const unsigned*)(vp + (size_t)kb * Dh);
      for (int i = tid; i < 2048; i += 128) {
        int key = i >> 5, dhp = (i & 31) << 1;
        unsigned u = src[i];
        vt[dhp * 64 + key]       = (unsigned short)(u & 0xFFFFu);
        vt[(dhp + 1) * 64 + key] = (unsigned short)(u >> 16);
      }
    }
    __builtin_prefetch(kp + (size_t)(kb + 64) * Dh, 0, 3);
    __builtin_prefetch(vp + (size_t)(kb + 64) * Dh, 0, 3);
    __syncthreads();

    // S = Q @ K^T for 4 key tiles (B frag = one 32B contiguous run per lane)
    v8f st[4];
#pragma unroll
    for (int t = 0; t < 4; ++t) {
      v8f acc = vzero;
#pragma unroll
      for (int cch = 0; cch < 2; ++cch) {
        Frag kf;
        const unsigned short* kr = &kt[(t * 16 + n) * 64 + cch * 32 + hh * 16];
        kf.q[0] = *(const u32x4*)kr;
        kf.q[1] = *(const u32x4*)(kr + 8);
        acc = wmma_bf16(qa[cch].v, kf.v, acc);
      }
      st[t] = acc;
    }

    // online softmax stats (row = r + 8*lanehalf; reduce within 16-lane half)
    float alpha[8], bs[8];
#pragma unroll
    for (int r = 0; r < 8; ++r) {
      float v0 = fmaxf(fmaxf(st[0][r], st[1][r]), fmaxf(st[2][r], st[3][r])) * SCALE;
      v0 = fmaxf(v0, __shfl_xor(v0, 1, 16));
      v0 = fmaxf(v0, __shfl_xor(v0, 2, 16));
      v0 = fmaxf(v0, __shfl_xor(v0, 4, 16));
      v0 = fmaxf(v0, __shfl_xor(v0, 8, 16));
      float mnew = fmaxf(mrun[r], v0);
      alpha[r] = __expf(mrun[r] - mnew);
      mrun[r] = mnew;
      bs[r] = 0.0f;
    }
#pragma unroll
    for (int t = 0; t < 4; ++t)
#pragma unroll
      for (int r = 0; r < 8; ++r) {
        float p = __expf(st[t][r] * SCALE - mrun[r]);
        bs[r] += p;
        pw[(r + 8 * hh) * 64 + t * 16 + n] = f32_to_bf16_bits(p);
      }
#pragma unroll
    for (int r = 0; r < 8; ++r) {
      float s = bs[r];
      s += __shfl_xor(s, 1, 16);
      s += __shfl_xor(s, 2, 16);
      s += __shfl_xor(s, 4, 16);
      s += __shfl_xor(s, 8, 16);
      lrun[r] = lrun[r] * alpha[r] + s;
    }
#pragma unroll
    for (int dt = 0; dt < 4; ++dt)
#pragma unroll
      for (int r = 0; r < 8; ++r) o[dt][r] *= alpha[r];

    // P back from LDS in A layout (same-wave RAW: explicit dscnt wait)
    asm volatile("s_wait_dscnt 0" ::: "memory");
    Frag pa[2];
#pragma unroll
    for (int cch = 0; cch < 2; ++cch) {
      const unsigned short* pr = &pw[n * 64 + cch * 32 + hh * 16];
      pa[cch].q[0] = *(const u32x4*)pr;
      pa[cch].q[1] = *(const u32x4*)(pr + 8);
    }

    // O += P @ V  (V B-frags are contiguous ds_load_b128 from transposed LDS)
#pragma unroll
    for (int dt = 0; dt < 4; ++dt) {
#pragma unroll
      for (int cch = 0; cch < 2; ++cch) {
        Frag vf;
        const unsigned short* vr = &vt[(dt * 16 + n) * 64 + cch * 32 + hh * 16];
        vf.q[0] = *(const u32x4*)vr;
        vf.q[1] = *(const u32x4*)(vr + 8);
        o[dt] = wmma_bf16(pa[cch].v, vf.v, o[dt]);
      }
    }
  }

  // finalize: O / l, store bf16 to attn layout [B,N,C]
#pragma unroll
  for (int r = 0; r < 8; ++r) {
    float inv = 1.0f / lrun[r];
    int token = qtile * 16 + r + 8 * hh;
    size_t row = ((size_t)b * N + token) * C + h * Dh;
#pragma unroll
    for (int dt = 0; dt < 4; ++dt)
      attn_ws[row + dt * 16 + n] = f32_to_bf16_bits(o[dt][r] * inv);
  }
}

// ---------------------------------------------------------------------------
// Kernel 3: output projection (bf16 WMMA) + bias, f32 out.
// ---------------------------------------------------------------------------
__global__ __launch_bounds__(128) void proj_kernel(
    const unsigned short* __restrict__ attn_ws,
    const unsigned short* __restrict__ wb,
    const float* __restrict__ proj_b,
    float* __restrict__ out)
{
  __shared__ unsigned short at[16 * C];          // 32 KB bf16 A tile
  const int tid = threadIdx.x, lane = tid & 31, wave = tid >> 5;
  const int mbase = blockIdx.x * 16;

  {
    const u32x4* src = (const u32x4*)(attn_ws + (size_t)mbase * C);
    u32x4* dst = (u32x4*)at;
    for (int i = tid; i < (16 * C) / 8; i += 128) dst[i] = src[i];
  }
  __syncthreads();

  const int jpair = blockIdx.y * 4 + wave;       // 0..31
  const int j0 = jpair * 32;
  const int n = lane & 15, hh = lane >> 4;
  const unsigned short* w0 = wb + (size_t)(j0 + n) * C;
  const unsigned short* w1 = wb + (size_t)(j0 + 16 + n) * C;

  v8f c0 = {}, c1 = {};
  for (int kc = 0; kc < C; kc += 32) {
    Frag a, b0, b1;
    const unsigned short* ar = &at[n * C + kc + hh * 16];
    a.q[0] = *(const u32x4*)ar;
    a.q[1] = *(const u32x4*)(ar + 8);
    const unsigned short* p0 = &w0[kc + hh * 16];
    b0.q[0] = *(const u32x4*)p0;
    b0.q[1] = *(const u32x4*)(p0 + 8);
    const unsigned short* p1 = &w1[kc + hh * 16];
    b1.q[0] = *(const u32x4*)p1;
    b1.q[1] = *(const u32x4*)(p1 + 8);
    __builtin_prefetch(&w0[kc + 32], 0, 3);
    c0 = wmma_bf16(a.v, b0.v, c0);
    c1 = wmma_bf16(a.v, b1.v, c1);
  }

  float bias0 = proj_b[j0 + n], bias1 = proj_b[j0 + 16 + n];
#pragma unroll
  for (int r = 0; r < 8; ++r) {
    int row = mbase + r + 8 * hh;
    out[(size_t)row * C + j0 + n]      = c0[r] + bias0;
    out[(size_t)row * C + j0 + 16 + n] = c1[r] + bias1;
  }
}

// ---------------------------------------------------------------------------
extern "C" void kernel_launch(void* const* d_in, const int* in_sizes, int n_in,
                              void* d_out, int out_size, void* d_ws, size_t ws_size,
                              hipStream_t stream) {
  (void)in_sizes; (void)n_in; (void)out_size; (void)ws_size;
  const float* x         = (const float*)d_in[0];
  const int*   positions = (const int*)d_in[1];
  const float* qkv_w     = (const float*)d_in[2];
  const float* proj_w    = (const float*)d_in[3];
  const float* proj_b    = (const float*)d_in[4];
  float* out = (float*)d_out;

  const size_t per = (size_t)B * H * N * Dh;     // 4M bf16 elems
  unsigned short* q_ws    = (unsigned short*)d_ws;
  unsigned short* k_ws    = q_ws + per;
  unsigned short* v_ws    = k_ws + per;
  unsigned short* attn_ws = v_ws + per;                 // [B,N,C] bf16
  unsigned short* xb      = attn_ws + (size_t)M * C;    // [M,C] bf16
  unsigned short* wqkvb   = xb + (size_t)M * C;         // [3C,C] bf16
  unsigned short* wprojb  = wqkvb + (size_t)JQ * C;     // [C,C] bf16

  // one-shot f32 -> bf16 conversion passes (L2-resident afterwards)
  cvt_bf16_kernel<<<(M * C / 4 + 255) / 256, 256, 0, stream>>>(x, xb, M * C / 4);
  cvt_bf16_kernel<<<(JQ * C / 4 + 255) / 256, 256, 0, stream>>>(qkv_w, wqkvb,
                                                                JQ * C / 4);
  cvt_bf16_kernel<<<(C * C / 4 + 255) / 256, 256, 0, stream>>>(proj_w, wprojb,
                                                               C * C / 4);

  qkv_rope_kernel<<<dim3(M / 16, JQ / 32 / 4), 128, 0, stream>>>(
      xb, positions, wqkvb, q_ws, k_ws, v_ws);
  flash_attn_kernel<<<dim3(N / 16 / 4, B * H), 128, 0, stream>>>(
      q_ws, k_ws, v_ws, attn_ws);
  proj_kernel<<<dim3(M / 16, C / 32 / 4), 128, 0, stream>>>(
      attn_ws, wprojb, proj_b, out);
}
